// GNN_11484742549910
// MI455X (gfx1250) — matrix operand order repaired and checked by reading, not compiled, not run on hardware
//
#include <hip/hip_runtime.h>

typedef __bf16 bf16_t;
typedef __attribute__((ext_vector_type(16))) __bf16 v16bf;
typedef __attribute__((ext_vector_type(8)))  __bf16 v8bf;
typedef __attribute__((ext_vector_type(8)))  float  v8f;
typedef __attribute__((ext_vector_type(4)))  unsigned int v4u;
typedef __attribute__((ext_vector_type(8)))  int v8i;
typedef __attribute__((ext_vector_type(4)))  int v4i;

#define GCN_N 50000
#define GCN_E 800000
#define GCN_FIN 128
#define GCN_FH 256

// ---------------- utility kernels ----------------

__global__ void gcn_cvt_bf16(const float* __restrict__ in, bf16_t* __restrict__ out, int n) {
    int i = blockIdx.x * blockDim.x + threadIdx.x;
    if (i < n) out[i] = (bf16_t)in[i];
}

// W is (K x Nn) row-major fp32; produce Wt (Nn x K) row-major bf16 so that
// B-fragments are contiguous in K.
__global__ void gcn_wt_cvt(const float* __restrict__ W, bf16_t* __restrict__ Wt, int K, int Nn) {
    int i = blockIdx.x * blockDim.x + threadIdx.x;
    if (i < K * Nn) {
        int k = i / Nn, n = i % Nn;
        Wt[n * K + k] = (bf16_t)W[i];
    }
}

__global__ void gcn_deg_init(float* __restrict__ deg, int n) {
    int i = blockIdx.x * blockDim.x + threadIdx.x;
    if (i < n) deg[i] = 1.0f;            // self-loop
}

__global__ void gcn_deg_accum(const long long* __restrict__ dst, float* __restrict__ deg, int e) {
    int i = blockIdx.x * blockDim.x + threadIdx.x;
    if (i < e) unsafeAtomicAdd(&deg[(int)dst[i]], 1.0f);
}

__global__ void gcn_deg_inv(const float* __restrict__ deg, float* __restrict__ dinv, int n) {
    int i = blockIdx.x * blockDim.x + threadIdx.x;
    if (i < n) dinv[i] = rsqrtf(deg[i]);
}

// ---------------- TDM tile load ----------------
// DMA a (rows x 32) bf16 tile (viewed as 8 x rows units of 8B) from global to
// LDS.  Row stride in LDS is padded to 80B via the descriptor's pad fields
// (16 DWORDs data + 4 DWORDs pad) -> bank-conflict-free b128 fragment reads.
// Rows beyond tensor_dim1 are zero-filled by hardware (handles the M tail).
__device__ static inline void gcn_tdm_load_tile(unsigned lds_off, const bf16_t* gptr,
                                                int rows_rem, int K /*row stride, elems*/) {
    unsigned long long ga = (unsigned long long)(size_t)gptr;
    unsigned td1 = (unsigned)(rows_rem > 128 ? 128 * 1024 : (rows_rem < 0 ? 0 : rows_rem));
    // (td1 just needs to be >= valid rows; cap keeps it in range for interior blocks)
    v4u g0;
    g0[0] = 1u;                                             // count = 1 descriptor
    g0[1] = lds_off;                                        // LDS byte address
    g0[2] = (unsigned)(ga & 0xffffffffu);                   // global_addr[31:0]
    g0[3] = (unsigned)((ga >> 32) & 0x01ffffffu) | (2u << 30); // addr[56:32] | type=2
    const unsigned td0     = 8u;                            // tensor_dim0 (8B units)
    const unsigned stride0 = (unsigned)(K >> 2);            // K*2B / 8B
    v8i g1;
    g1[0] = (int)((3u << 16)      // data_size = 8B
                | (1u << 20)      // pad_enable
                | (3u << 22)      // pad_interval: 16 DWORDs (64B) between pads
                | (3u << 25));    // pad_amount: 4 DWORDs (16B)
    g1[1] = (int)((td0 & 0xffffu) << 16);                            // tensor_dim0[15:0]
    g1[2] = (int)(((td0 >> 16) & 0xffffu) | ((td1 & 0xffffu) << 16));// td0 hi | td1 lo
    g1[3] = (int)(((td1 >> 16) & 0xffffu) | (8u << 16));             // td1 hi | tile_dim0=8
    g1[4] = (int)(128u);                                             // tile_dim1=128, tile_dim2=0
    g1[5] = (int)stride0;                                            // dim0_stride[31:0]
    g1[6] = 0;                                                       // stride hi | dim1_stride lo
    g1[7] = 0;
    v4i z4 = {0, 0, 0, 0};
#if defined(__clang_major__) && (__clang_major__ >= 23)
    v8i z8 = {0, 0, 0, 0, 0, 0, 0, 0};
    __builtin_amdgcn_tensor_load_to_lds(g0, g1, z4, z4, z8, 0);
#else
    __builtin_amdgcn_tensor_load_to_lds(g0, g1, z4, z4, 0);
#endif
}

__device__ static inline unsigned gcn_lds_off(const void* p) {
    // flat->LDS mapping truncates to low 32 bits (ISA 10.2 aperture table)
    return (unsigned)(unsigned long long)(size_t)p;
}

// ---------------- bf16 WMMA GEMM, TDM-fed, double-buffered ----------------
// Y = H(MxK) @ W(KxN), N == 256.  Writes Ys[r][c] = dinv[r]*Y[r][c] and
// Agg[r][c] = same value (self-loop init for the scatter-add).
// Block: 256 threads = 8 wave32.  Tile: BM=128, BN=128, BK=32.
// Wave w computes columns [n0+16w, n0+16w+16) over 8 M-subtiles of 16.

__global__ __launch_bounds__(256)
void gcn_gemm(const bf16_t* __restrict__ H, const bf16_t* __restrict__ Wt,
              const float* __restrict__ dinv, float* __restrict__ Ys,
              float* __restrict__ Agg, int M, int K) {
    __shared__ __align__(16) bf16_t As[2][128][40];   // 80B padded rows (TDM pad fields)
    __shared__ __align__(16) bf16_t Bs[2][128][40];

    const int t    = threadIdx.x;
    const int m0   = blockIdx.x * 128;
    const int n0   = blockIdx.y * 128;
    const int lane = t & 31;
    const int wave = t >> 5;
    const int half = lane >> 4;
    const int lr   = lane & 15;
    const bool issuer = (t < 32);     // wave 0 drives the TDM

    v8f acc[8];
#pragma unroll
    for (int i = 0; i < 8; ++i) {
#pragma unroll
        for (int j = 0; j < 8; ++j) acc[i][j] = 0.0f;
    }

    const int nk = K >> 5;
    if (issuer) {
        gcn_tdm_load_tile(gcn_lds_off(&As[0][0][0]), H + (size_t)m0 * K, M - m0, K);
        gcn_tdm_load_tile(gcn_lds_off(&Bs[0][0][0]), Wt + (size_t)n0 * K, 128, K);
    }

    for (int kt = 0; kt < nk; ++kt) {
        const int cur = kt & 1;
        const int nxt = cur ^ 1;
        if (issuer) {
            if (kt + 1 < nk) {
                const int k0n = (kt + 1) << 5;
                gcn_tdm_load_tile(gcn_lds_off(&As[nxt][0][0]),
                                  H + (size_t)m0 * K + k0n, M - m0, K);
                gcn_tdm_load_tile(gcn_lds_off(&Bs[nxt][0][0]),
                                  Wt + (size_t)n0 * K + k0n, 128, K);
                __builtin_amdgcn_s_wait_tensorcnt((short)2);  // current pair done
            } else {
                __builtin_amdgcn_s_wait_tensorcnt((short)0);
            }
        }
        __syncthreads();   // TDM data visible to all waves

        // B fragment (32x16 bf16): lane holds col lr, K = 16*half + 0..15
        v8bf blo = *(const v8bf*)&Bs[cur][16 * wave + lr][16 * half];
        v8bf bhi = *(const v8bf*)&Bs[cur][16 * wave + lr][16 * half + 8];
        v16bf bfrag;
#pragma unroll
        for (int j = 0; j < 8; ++j) { bfrag[j] = blo[j]; bfrag[j + 8] = bhi[j]; }

        // Load all 8 A fragments first so the WMMAs issue back-to-back
        v16bf afrag[8];
#pragma unroll
        for (int mt = 0; mt < 8; ++mt) {
            v8bf alo = *(const v8bf*)&As[cur][16 * mt + lr][16 * half];
            v8bf ahi = *(const v8bf*)&As[cur][16 * mt + lr][16 * half + 8];
#pragma unroll
            for (int j = 0; j < 8; ++j) { afrag[mt][j] = alo[j]; afrag[mt][j + 8] = ahi[j]; }
        }
#pragma unroll
        for (int mt = 0; mt < 8; ++mt) {
            acc[mt] = __builtin_amdgcn_wmma_f32_16x16x32_bf16(
                false, afrag[mt], false, bfrag, (short)0, acc[mt], false, false);
        }

        __syncthreads();   // everyone done with buf 'cur' before it is re-DMAed
    }

    // Epilogue: C layout -> lane holds col (lr), rows 8*half + i in VGPR i.
    const int col = n0 + 16 * wave + lr;
#pragma unroll
    for (int mt = 0; mt < 8; ++mt) {
        int r0 = m0 + 16 * mt + 8 * half;
#pragma unroll
        for (int i = 0; i < 8; ++i) {
            int r = r0 + i;
            if (r < M) {
                float v = acc[mt][i] * dinv[r];
                size_t off = (size_t)r * GCN_FH + col;
                Ys[off]  = v;   // gather source for edge scatter
                Agg[off] = v;   // self-loop contribution initializes the sum
            }
        }
    }
}

// ---------------- edge scatter: Agg[dst] += Ys[src] ----------------
// 64 threads (float4 chunks) per edge; fp32 hardware atomics.
__global__ void gcn_scatter(const long long* __restrict__ src, const long long* __restrict__ dst,
                            const float* __restrict__ Ys, float* __restrict__ Agg) {
    long long g = (long long)blockIdx.x * blockDim.x + threadIdx.x;
    int e = (int)(g >> 6);
    if (e >= GCN_E) return;
    int c = ((int)g & 63) << 2;
    int s = (int)src[e];
    int d = (int)dst[e];
    const float4 v = *(const float4*)(Ys + (size_t)s * GCN_FH + c);
    float* o = Agg + (size_t)d * GCN_FH + c;
    unsafeAtomicAdd(o + 0, v.x);
    unsafeAtomicAdd(o + 1, v.y);
    unsafeAtomicAdd(o + 2, v.z);
    unsafeAtomicAdd(o + 3, v.w);
}

// ---------------- finalize: dinv[dst] post-scale + bias (+relu) ----------------

__global__ void gcn_finalize_relu(const float* __restrict__ Agg, const float* __restrict__ bias,
                                  const float* __restrict__ dinv, bf16_t* __restrict__ Hn, int M) {
    int g = blockIdx.x * blockDim.x + threadIdx.x;   // over M*64 float4 chunks
    int r = g >> 6;
    if (r >= M) return;
    int c4 = (g & 63) << 2;
    float4 a = *(const float4*)(Agg + ((size_t)g << 2));
    float4 b = *(const float4*)(bias + c4);
    float di = dinv[r];
    union { bf16_t h[4]; uint2 u; } o;
    o.h[0] = (bf16_t)fmaxf(di * a.x + b.x, 0.0f);
    o.h[1] = (bf16_t)fmaxf(di * a.y + b.y, 0.0f);
    o.h[2] = (bf16_t)fmaxf(di * a.z + b.z, 0.0f);
    o.h[3] = (bf16_t)fmaxf(di * a.w + b.w, 0.0f);
    ((uint2*)Hn)[g] = o.u;
}

__global__ void gcn_finalize_out(const float* __restrict__ Agg, const float* __restrict__ bias,
                                 const float* __restrict__ dinv, float* __restrict__ out, int M) {
    int g = blockIdx.x * blockDim.x + threadIdx.x;   // over M*64 float4 chunks
    int r = g >> 6;
    if (r >= M) return;
    int c4 = (g & 63) << 2;
    float4 a = *(const float4*)(Agg + ((size_t)g << 2));
    float4 b = *(const float4*)(bias + c4);
    float di = dinv[r];
    float4 o;
    o.x = di * a.x + b.x;
    o.y = di * a.y + b.y;
    o.z = di * a.z + b.z;
    o.w = di * a.w + b.w;
    *(float4*)(out + ((size_t)g << 2)) = o;
}

// ---------------- host launch ----------------

extern "C" void kernel_launch(void* const* d_in, const int* in_sizes, int n_in,
                              void* d_out, int out_size, void* d_ws, size_t ws_size,
                              hipStream_t stream) {
    const float*     x  = (const float*)d_in[0];
    const long long* ei = (const long long*)d_in[1];   // [2, E] int64
    const float* W1 = (const float*)d_in[2];
    const float* b1 = (const float*)d_in[3];
    const float* W2 = (const float*)d_in[4];
    const float* b2 = (const float*)d_in[5];
    const float* W3 = (const float*)d_in[6];
    const float* b3 = (const float*)d_in[7];
    float* out = (float*)d_out;

    const int N = GCN_N, E = GCN_E;

    // workspace carve-out (all 256B aligned)
    char* ws = (char*)d_ws;
    auto carve = [&](size_t bytes) -> char* {
        char* p = ws;
        ws += (bytes + 255) & ~(size_t)255;
        return p;
    };
    float*  deg  = (float*)carve((size_t)N * 4);
    float*  dinv = (float*)carve((size_t)N * 4);
    bf16_t* Wt1  = (bf16_t*)carve((size_t)GCN_FIN * GCN_FH * 2);
    bf16_t* Wt2  = (bf16_t*)carve((size_t)GCN_FH * GCN_FH * 2);
    bf16_t* Wt3  = (bf16_t*)carve((size_t)GCN_FH * GCN_FH * 2);
    bf16_t* Ha   = (bf16_t*)carve((size_t)N * GCN_FH * 2);
    bf16_t* Hb   = (bf16_t*)carve((size_t)N * GCN_FH * 2);
    float*  Ys   = (float*)carve((size_t)N * GCN_FH * 4);
    float*  Agg  = (float*)carve((size_t)N * GCN_FH * 4);

    const int TB = 256;
    const long long sthreads = (long long)E * 64;
    const int sblocks = (int)((sthreads + TB - 1) / TB);
    const int fblocks = (N * 64 + TB - 1) / TB;        // float4-vectorized finalize
    dim3 gemm_grid((N + 127) / 128, 2);

    // Precompute bf16 inputs / transposed weights
    gcn_cvt_bf16<<<(N * GCN_FIN + TB - 1) / TB, TB, 0, stream>>>(x, Ha, N * GCN_FIN);
    gcn_wt_cvt<<<(GCN_FIN * GCN_FH + TB - 1) / TB, TB, 0, stream>>>(W1, Wt1, GCN_FIN, GCN_FH);
    gcn_wt_cvt<<<(GCN_FH * GCN_FH + TB - 1) / TB, TB, 0, stream>>>(W2, Wt2, GCN_FH, GCN_FH);
    gcn_wt_cvt<<<(GCN_FH * GCN_FH + TB - 1) / TB, TB, 0, stream>>>(W3, Wt3, GCN_FH, GCN_FH);

    // Degrees (over dst incl. self-loop) and dinv = deg^{-1/2}
    gcn_deg_init<<<(N + TB - 1) / TB, TB, 0, stream>>>(deg, N);
    gcn_deg_accum<<<(E + TB - 1) / TB, TB, 0, stream>>>(ei + E, deg, E);
    gcn_deg_inv<<<(N + TB - 1) / TB, TB, 0, stream>>>(deg, dinv, N);

    // Layer 1: K = 128
    gcn_gemm<<<gemm_grid, TB, 0, stream>>>(Ha, Wt1, dinv, Ys, Agg, N, GCN_FIN);
    gcn_scatter<<<sblocks, TB, 0, stream>>>(ei, ei + E, Ys, Agg);
    gcn_finalize_relu<<<fblocks, TB, 0, stream>>>(Agg, b1, dinv, Hb, N);

    // Layer 2: K = 256
    gcn_gemm<<<gemm_grid, TB, 0, stream>>>(Hb, Wt2, dinv, Ys, Agg, N, GCN_FH);
    gcn_scatter<<<sblocks, TB, 0, stream>>>(ei, ei + E, Ys, Agg);
    gcn_finalize_relu<<<fblocks, TB, 0, stream>>>(Agg, b2, dinv, Ha, N);

    // Layer 3: K = 256, fp32 output, no relu
    gcn_gemm<<<gemm_grid, TB, 0, stream>>>(Ha, Wt3, dinv, Ys, Agg, N, GCN_FH);
    gcn_scatter<<<sblocks, TB, 0, stream>>>(ei, ei + E, Ys, Agg);
    gcn_finalize_out<<<fblocks, TB, 0, stream>>>(Agg, b3, dinv, out, N);
}
